// MyYolo_70394513982008
// MI455X (gfx1250) — matrix operand order repaired and checked by reading, not compiled, not run on hardware
//
#include <hip/hip_runtime.h>

typedef __attribute__((ext_vector_type(16))) _Float16 v16h;
typedef __attribute__((ext_vector_type(8)))  _Float16 v8h;
typedef __attribute__((ext_vector_type(8)))  float    v8f;

#define KCI 256  // ci-chunk staged in LDS (multiple of 32)
#define MT  4    // 16-channel subtiles per block -> 64 output channels/block
#define NT  2    // 16-pixel column tiles per wave  -> 32 pixels/wave

// --------------------------------------------------------------- utilities
// NCHW f32 -> NHWC f16 input conversion
__global__ void k_nchw2nhwc(const float* __restrict__ X, _Float16* __restrict__ Y,
                            int Bn, int C, int H, int W) {
  long i = (long)blockIdx.x * blockDim.x + threadIdx.x;
  long total = (long)Bn * C * H * W;
  long st = (long)gridDim.x * blockDim.x;
  for (; i < total; i += st) {
    int c = (int)(i % C); long t = i / C;
    int x = (int)(t % W); t /= W;
    int y = (int)(t % H); int b = (int)(t / H);
    Y[i] = (_Float16)X[(((long)b * C + c) * H + y) * W + x];
  }
}

// Fold BN into weights and reorder OIHW -> [O][ky*kx][CinP] (ci padded to 32),
// f16. bias = b - m*scale (or passthrough bias for convs without BN).
__global__ void k_prep(const float* __restrict__ w,
                       const float* __restrict__ g, const float* __restrict__ b,
                       const float* __restrict__ m, const float* __restrict__ v,
                       const float* __restrict__ bias_in,
                       _Float16* __restrict__ wh, float* __restrict__ bias,
                       int Cout, int Cin, int KHW, int CinP) {
  long i0 = (long)blockIdx.x * blockDim.x + threadIdx.x;
  if (i0 < Cout) {
    float s = 1.f, t = 0.f;
    if (g) { s = g[i0] * rsqrtf(v[i0] + 1e-5f); t = b[i0] - m[i0] * s; }
    else if (bias_in) t = bias_in[i0];
    bias[i0] = t;
  }
  const long KdimP = (long)KHW * CinP;
  long total = (long)Cout * KdimP;
  long st = (long)gridDim.x * blockDim.x;
  for (long i = i0; i < total; i += st) {
    int o = (int)(i / KdimP); long rem = i - (long)o * KdimP;
    int t = (int)(rem / CinP); int ci = (int)(rem - (long)t * CinP);
    float val = 0.f;
    if (ci < Cin) {
      float s = g ? g[o] * rsqrtf(v[o] + 1e-5f) : 1.f;
      val = w[((long)o * Cin + ci) * KHW + t] * s;
    }
    wh[i] = (_Float16)val;
  }
}

// ------------------------------------------- implicit-GEMM conv with WMMA
// NHWC activations, weights [Cout][KHW][CinP] -> GEMM K ordered (ky,kx,ci).
// Block: 256 threads = 8 waves; each wave owns 64(M) x 32(N) as MT*NT=8
// accumulators. Per 32-wide K step: 2 B fragments (2x b128 NHWC loads each),
// 4 A fragments (2x ds_load_b128 each) reused across both pixel columns ->
// 8 V_WMMA_F32_16X16X32_F16 per 8 ds ops. No integer division in hot loop.
// act: 0=none 1=relu 2=leaky(0.1). Res != null -> residual add before act.
__global__ void __launch_bounds__(256) k_conv_wmma(
    const _Float16* __restrict__ X, const _Float16* __restrict__ Wt,
    const float* __restrict__ bias, _Float16* __restrict__ Y,
    const _Float16* __restrict__ Res,
    int Cin, int CinP, int H, int W, int Cout, int KKs,
    int stride, int pad, int Ho, int Wo, int Npix, int act)
{
  __shared__ _Float16 sA[16 * MT * KCI];          // 32 KB weight panel
  const int KHW = KKs * KKs;
  const long KdimP = (long)KHW * CinP;
  const int tid  = threadIdx.x;
  const int lane = tid & 31;
  const int wave = tid >> 5;
  const int M0   = blockIdx.y * (16 * MT);
  const int n16  = lane & 15;
  const int hi   = lane >> 4;                     // lane half (ISA 16/16 split)
  const int kb8  = hi << 3;
  const int HoWo = Ho * Wo;
  const bool vecB = ((Cin & 15) == 0);

  int pj[NT], bI[NT], oyj[NT], oxj[NT];
#pragma unroll
  for (int j = 0; j < NT; ++j) {
    const int pp = (blockIdx.x << 8) + (wave << 5) + (j << 4) + n16;
    pj[j] = pp;
    const int ps = pp < Npix ? pp : Npix - 1;
    const int bb = ps / HoWo;
    const int rem = ps - bb * HoWo;
    bI[j] = bb;
    oyj[j] = rem / Wo;
    oxj[j] = rem - oyj[j] * Wo;
  }

  v8f acc[NT][MT];
#pragma unroll
  for (int j = 0; j < NT; ++j)
#pragma unroll
    for (int s = 0; s < MT; ++s) acc[j][s] = {};

  for (int t = 0; t < KHW; ++t) {                 // outer: kernel position
    const int ky = t / KKs;
    const int kx = t - ky * KKs;
    bool inb[NT];
    const _Float16* Xp[NT];
#pragma unroll
    for (int j = 0; j < NT; ++j) {
      const int iy = oyj[j] * stride - pad + ky;
      const int ix = oxj[j] * stride - pad + kx;
      inb[j] = (iy >= 0 && iy < H && ix >= 0 && ix < W);
      Xp[j] = X + (((long)bI[j] * H + (inb[j] ? iy : 0)) * W + (inb[j] ? ix : 0)) * Cin;
    }
    const _Float16* Wp = Wt + (long)t * CinP;

    for (int c0 = 0; c0 < CinP; c0 += KCI) {
      const int kcEnd = (CinP - c0) < KCI ? (CinP - c0) : KCI;
      __syncthreads();
      // vectorized stage: 64 rows x kcEnd cols, 4 threads/row, b128 copies
      {
        const int mm = tid >> 2;
        const int q  = (tid & 3) << 3;
        const int oc = M0 + mm;
        for (int kk = q; kk < kcEnd; kk += 32) {
          v8h wv = {};
          if (oc < Cout)
            wv = *(const v8h*)(Wp + (long)oc * KdimP + c0 + kk);
          *(v8h*)&sA[(mm << 8) + kk] = wv;
        }
      }
      if (c0 + KCI < CinP)                        // pull next weight chunk
        __builtin_prefetch(Wp + (long)M0 * KdimP + c0 + KCI, 0, 0);
      __syncthreads();

      for (int cc = 0; cc < kcEnd; cc += 32) {
        // ---- NT B fragments: 16 consecutive ci values per lane (NHWC) ----
        const int cbase = c0 + cc + (hi << 4);
        union U { v16h v; v8h h[2]; };
        U bu[NT];
#pragma unroll
        for (int j = 0; j < NT; ++j) {
          if (inb[j] && vecB && (cbase + 16 <= Cin)) {
            const v8h* pb = (const v8h*)(Xp[j] + cbase);
            bu[j].h[0] = pb[0];
            bu[j].h[1] = pb[1];
          } else {
#pragma unroll
            for (int e = 0; e < 16; ++e) {
              int ci = cbase + e;
              bu[j].v[e] = (inb[j] && ci < Cin) ? Xp[j][ci] : (_Float16)0.f;
            }
          }
        }
        // ---- 4 A fragments (2x ds_load_b128 each), each feeds NT WMMAs ----
#pragma unroll
        for (int s = 0; s < MT; ++s) {
          U au;
          const int ro = (((s << 4) + n16) << 8) + cc + (hi << 4);
          au.h[0] = *(const v8h*)&sA[ro];
          au.h[1] = *(const v8h*)&sA[ro + 8];
#pragma unroll
          for (int j = 0; j < NT; ++j)
            acc[j][s] = __builtin_amdgcn_wmma_f32_16x16x32_f16(false, au.v, false, bu[j].v,
                                                               (short)0, acc[j][s], false, false);
        }
      }
    }
  }

  // epilogue: bias + residual + activation, packed NHWC stores
#pragma unroll
  for (int j = 0; j < NT; ++j) {
    if (pj[j] < Npix) {
      const long obase = (((long)bI[j] * Ho + oyj[j]) * Wo + oxj[j]) * Cout;
#pragma unroll
      for (int s = 0; s < MT; ++s) {
        const int oc0 = M0 + (s << 4) + kb8;      // 8 consecutive channels
        if (oc0 < Cout) {
          if (((Cout & 7) == 0) && (oc0 + 8 <= Cout)) {
            float rr[8];
            if (Res) {
              v8h r8 = *(const v8h*)(Res + obase + oc0);
#pragma unroll
              for (int f = 0; f < 8; ++f) rr[f] = (float)r8[f];
            }
            v8h o8;
#pragma unroll
            for (int f = 0; f < 8; ++f) {
              float vv = acc[j][s][f] + bias[oc0 + f];
              if (Res) vv += rr[f];
              if (act == 1)      vv = vv > 0.f ? vv : 0.f;
              else if (act == 2) vv = vv > 0.f ? vv : 0.1f * vv;
              o8[f] = (_Float16)vv;
            }
            *(v8h*)(Y + obase + oc0) = o8;
          } else {
#pragma unroll
            for (int f = 0; f < 8; ++f) {
              const int oc = oc0 + f;
              if (oc < Cout) {
                float vv = acc[j][s][f] + bias[oc];
                if (Res) vv += (float)Res[obase + oc];
                if (act == 1)      vv = vv > 0.f ? vv : 0.f;
                else if (act == 2) vv = vv > 0.f ? vv : 0.1f * vv;
                Y[obase + oc] = (_Float16)vv;
              }
            }
          }
        }
      }
    }
  }
}

// ------------------------------------------------------------- maxpool/copy
// NHWC. Writes into output with Ctot channels at offset Coff (SPP concat);
// k=1 acts as channel-slice copy.
__global__ void k_maxpool(const _Float16* __restrict__ X, _Float16* __restrict__ Y,
                          int Bn, int C, int H, int W,
                          int k, int s, int pad, int Ho, int Wo,
                          int Ctot, int Coff) {
  long i = (long)blockIdx.x * blockDim.x + threadIdx.x;
  long total = (long)Bn * C * Ho * Wo;
  if (i >= total) return;
  int c = (int)(i % C); long t = i / C;
  int ox = (int)(t % Wo); t /= Wo;
  int oy = (int)(t % Ho);
  int b  = (int)(t / Ho);
  float best = -3.4e38f;
  for (int ky = 0; ky < k; ++ky) {
    int iy = oy * s - pad + ky;
    if (iy < 0 || iy >= H) continue;
    for (int kx = 0; kx < k; ++kx) {
      int ix = ox * s - pad + kx;
      if (ix < 0 || ix >= W) continue;
      float v = (float)X[(((long)b * H + iy) * W + ix) * C + c];
      best = v > best ? v : best;
    }
  }
  Y[(((long)b * Ho + oy) * Wo + ox) * Ctot + Coff + c] = (_Float16)best;
}

// SAM gate: y = x * sigmoid(t)  (elementwise, layout agnostic)
__global__ void k_sam(const _Float16* __restrict__ X, const _Float16* __restrict__ T,
                      _Float16* __restrict__ Y, long n) {
  long i = (long)blockIdx.x * blockDim.x + threadIdx.x;
  if (i >= n) return;
  float t = (float)T[i];
  float sg = 1.f / (1.f + expf(-t));
  Y[i] = (_Float16)((float)X[i] * sg);
}

// ---------------------------------------------------------- YOLO box decode
// pred NHWC (B,16,16,85) f16 -> boxes/classes into d_out, scores to ws.
__global__ void k_decode(const _Float16* __restrict__ P, float* __restrict__ out,
                         float* __restrict__ ws_scores,
                         int Bn, int HW, int Wgrid, float stride_pix) {
  int i = blockIdx.x * blockDim.x + threadIdx.x;
  if (i >= Bn * HW) return;
  int p = i % HW;
  float v[85];
  const _Float16* Pp = P + (long)i * 85;          // channels contiguous
  for (int c = 0; c < 85; ++c) v[c] = (float)Pp[c];
  float conf = 1.f / (1.f + expf(-v[0]));
  float mx = -3.4e38f;
  for (int c = 1; c <= 80; ++c) mx = fmaxf(mx, v[c]);
  float sum = 0.f;
  for (int c = 1; c <= 80; ++c) sum += expf(v[c] - mx);
  float bestS = -1.f; int bestC = 0;
  for (int c = 0; c < 80; ++c) {
    float s = expf(v[1 + c] - mx) / sum * conf;
    if (s > bestS) { bestS = s; bestC = c; }
  }
  float score = bestS > 0.01f ? bestS : 0.f;
  int gy = p / Wgrid, gx = p % Wgrid;
  float cx = (1.f / (1.f + expf(-v[81])) + gx) * stride_pix;
  float cy = (1.f / (1.f + expf(-v[82])) + gy) * stride_pix;
  float t2 = fminf(fmaxf(v[83], -10.f), 10.f);
  float t3 = fminf(fmaxf(v[84], -10.f), 10.f);
  float w = expf(t2) * stride_pix, h = expf(t3) * stride_pix;
  float* boxes   = out;                           // [B*HW*4]
  float* classes = out + (long)Bn * HW * 5;       // after boxes + scores
  boxes[(long)i * 4 + 0] = cx - 0.5f * w;
  boxes[(long)i * 4 + 1] = cy - 0.5f * h;
  boxes[(long)i * 4 + 2] = cx + 0.5f * w;
  boxes[(long)i * 4 + 3] = cy + 0.5f * h;
  classes[i]   = (float)bestC;
  ws_scores[i] = score;
}

// ------------------------------------------------------------------- NMS
// One block per batch image. Class-offset boxes, stable descending sort,
// sequential suppression (matches reference lax.scan semantics).
__global__ void k_nms(const float* __restrict__ boxes, const float* __restrict__ classes,
                      const float* __restrict__ ws_scores, float* __restrict__ out_scores,
                      int HW) {
  __shared__ float x1[256], y1[256], x2[256], y2[256], sc[256];
  __shared__ int ord[256];
  __shared__ unsigned char keep[256];
  int b = blockIdx.x;
  int t = threadIdx.x;
  if (t < HW) {
    long i = (long)b * HW + t;
    float off = classes[i] * 1024.f;              // 2*IN per class
    x1[t] = boxes[i * 4 + 0] + off;
    y1[t] = boxes[i * 4 + 1] + off;
    x2[t] = boxes[i * 4 + 2] + off;
    y2[t] = boxes[i * 4 + 3] + off;
    sc[t] = ws_scores[i];
  }
  __syncthreads();
  if (t < HW) {                                   // rank = stable argsort(-score)
    int r = 0;
    for (int j = 0; j < HW; ++j)
      if (sc[j] > sc[t] || (sc[j] == sc[t] && j < t)) r++;
    ord[r] = t;
  }
  __syncthreads();
  if (t == 0) {
    for (int a = 0; a < HW; ++a) {
      int i = ord[a];
      bool kp = sc[i] > 0.f;
      if (kp) {
        float ax1 = x1[i], ay1 = y1[i], ax2 = x2[i], ay2 = y2[i];
        float areaA = (ax2 - ax1) * (ay2 - ay1);
        for (int c = 0; c < a && kp; ++c) {
          int j = ord[c];
          if (!keep[j]) continue;
          float iw = fminf(ax2, x2[j]) - fmaxf(ax1, x1[j]); iw = iw > 0.f ? iw : 0.f;
          float ih = fminf(ay2, y2[j]) - fmaxf(ay1, y1[j]); ih = ih > 0.f ? ih : 0.f;
          float inter = iw * ih;
          float areaB = (x2[j] - x1[j]) * (y2[j] - y1[j]);
          float iou = inter / (areaA + areaB - inter + 1e-9f);
          if (iou > 0.5f) kp = false;
        }
      }
      keep[i] = kp ? 1 : 0;
    }
    for (int i2 = 0; i2 < HW; ++i2)
      out_scores[(long)b * HW + i2] = keep[i2] ? sc[i2] : 0.f;
  }
}

// =========================================================== host orchestration
extern "C" void kernel_launch(void* const* d_in, const int* in_sizes, int n_in,
                              void* d_out, int out_size, void* d_ws, size_t ws_size,
                              hipStream_t stream) {
  (void)in_sizes; (void)n_in; (void)out_size; (void)ws_size;
  const int Bn = 4;

  int cur = 0;
  auto nextp = [&]() -> const float* { return (const float*)d_in[cur++]; };
  struct BNp { const float *g, *b, *m, *v; };
  auto nextbn = [&]() -> BNp { BNp r; r.g = nextp(); r.b = nextp(); r.m = nextp(); r.v = nextp(); return r; };

  size_t off = 0;
  char* base = (char*)d_ws;
  auto alloc = [&](size_t bytes) -> void* {
    void* r = base + off;
    off = (off + bytes + 255) & ~(size_t)255;
    return r;
  };

  struct CW { _Float16* w; float* bias; int Cout, Cin, CinP, K; };
  auto prep = [&](const float* w, const BNp* bn, const float* bias_in,
                  int Cout, int Cin, int K) -> CW {
    CW c; c.Cout = Cout; c.Cin = Cin; c.K = K;
    c.CinP = (Cin + 31) & ~31;
    int KHW = K * K;
    long KdimP = (long)KHW * c.CinP;
    c.w = (_Float16*)alloc((size_t)Cout * KdimP * sizeof(_Float16));
    c.bias = (float*)alloc((size_t)Cout * sizeof(float));
    long total = (long)Cout * KdimP;
    int blk = (int)((total + 255) / 256); if (blk > 8192) blk = 8192; if (blk < 1) blk = 1;
    k_prep<<<blk, 256, 0, stream>>>(w, bn ? bn->g : nullptr, bn ? bn->b : nullptr,
                                    bn ? bn->m : nullptr, bn ? bn->v : nullptr,
                                    bias_in, c.w, c.bias, Cout, Cin, KHW, c.CinP);
    return c;
  };

  auto conv = [&](const _Float16* X, const CW& cw, _Float16* Y, const _Float16* Res,
                  int H, int W, int stride, int pad, int act) {
    int Ho = (H + 2 * pad - cw.K) / stride + 1;
    int Wo = (W + 2 * pad - cw.K) / stride + 1;
    int Npix = Bn * Ho * Wo;
    dim3 g((unsigned)((Npix + 255) / 256), (unsigned)((cw.Cout + 16 * MT - 1) / (16 * MT)));
    k_conv_wmma<<<g, 256, 0, stream>>>(X, cw.w, cw.bias, Y, Res,
                                       cw.Cin, cw.CinP, H, W, cw.Cout, cw.K,
                                       stride, pad, Ho, Wo, Npix, act);
  };

  // ----- read params in setup_inputs() order, fold BN on device -----
  const float* data = nextp();
  const float* stem_w = nextp(); BNp stem_bn = nextbn();
  CW stem = prep(stem_w, &stem_bn, nullptr, 64, 3, 7);

  struct BlockCW { CW c1, c2, cd; bool down; };
  auto read_block = [&](int Cin, int Cout, bool down) -> BlockCW {
    BlockCW bl; bl.down = down;
    const float* w1 = nextp(); BNp bn1 = nextbn();
    const float* w2 = nextp(); BNp bn2 = nextbn();
    bl.c1 = prep(w1, &bn1, nullptr, Cout, Cin, 3);
    bl.c2 = prep(w2, &bn2, nullptr, Cout, Cout, 3);
    if (down) { const float* wd = nextp(); BNp bnd = nextbn(); bl.cd = prep(wd, &bnd, nullptr, Cout, Cin, 1); }
    return bl;
  };
  BlockCW l1a = read_block(64, 64, false);
  BlockCW l1b = read_block(64, 64, false);
  BlockCW l2a = read_block(64, 128, true);
  BlockCW l2b = read_block(128, 128, false);
  BlockCW l3a = read_block(128, 256, true);
  BlockCW l3b = read_block(256, 256, false);
  BlockCW l4a = read_block(256, 512, true);
  BlockCW l4b = read_block(512, 512, false);

  const float* spp_w = nextp(); BNp spp_bn = nextbn();
  CW spp = prep(spp_w, &spp_bn, nullptr, 512, 2048, 1);
  const float* sam_w = nextp(); const float* sam_b = nextp();
  CW sam = prep(sam_w, nullptr, sam_b, 512, 512, 1);
  CW cs[4];
  { int ci[4] = {512, 256, 512, 256}; int co[4] = {256, 512, 256, 512}; int kk[4] = {1, 3, 1, 3};
    for (int i = 0; i < 4; ++i) { const float* w = nextp(); BNp bn = nextbn(); cs[i] = prep(w, &bn, nullptr, co[i], ci[i], kk[i]); } }
  const float* pred_w = nextp(); const float* pred_b = nextp();
  CW pred = prep(pred_w, nullptr, pred_b, 85, 512, 1);

  // ----- forward pass (all activations f16, NHWC) -----
  auto szf = [&](int C, int Hh) { return (size_t)Bn * C * Hh * Hh * sizeof(_Float16); };

  _Float16* xh = (_Float16*)alloc(szf(3, 512));
  k_nchw2nhwc<<<4096, 256, 0, stream>>>(data, xh, Bn, 3, 512, 512);

  _Float16* a0 = (_Float16*)alloc(szf(64, 256));
  conv(xh, stem, a0, nullptr, 512, 512, 2, 3, 1);
  _Float16* x0 = (_Float16*)alloc(szf(64, 128));
  { long n = (long)Bn * 64 * 128 * 128;
    k_maxpool<<<(int)((n + 255) / 256), 256, 0, stream>>>(a0, x0, Bn, 64, 256, 256, 3, 2, 1, 128, 128, 64, 0); }

  auto run_block = [&](const BlockCW& bl, const _Float16* xin, int H, int stride,
                       _Float16* ybuf, _Float16* scbuf, _Float16* out) {
    int Ho = (H + 2 - 3) / stride + 1;
    conv(xin, bl.c1, ybuf, nullptr, H, H, stride, 1, 1);
    const _Float16* res = xin;
    if (bl.down) { conv(xin, bl.cd, scbuf, nullptr, H, H, stride, 0, 0); res = scbuf; }
    conv(ybuf, bl.c2, out, res, Ho, Ho, 1, 1, 1);
  };

  // layer1 @128, 64ch
  _Float16* y1 = (_Float16*)alloc(szf(64, 128));
  _Float16* o0 = (_Float16*)alloc(szf(64, 128));
  _Float16* o1 = (_Float16*)alloc(szf(64, 128));
  run_block(l1a, x0, 128, 1, y1, nullptr, o0);
  run_block(l1b, o0, 128, 1, y1, nullptr, o1);
  // layer2 -> @64, 128ch
  _Float16* y2  = (_Float16*)alloc(szf(128, 64));
  _Float16* sc2 = (_Float16*)alloc(szf(128, 64));
  _Float16* p0  = (_Float16*)alloc(szf(128, 64));
  _Float16* p1  = (_Float16*)alloc(szf(128, 64));
  run_block(l2a, o1, 128, 2, y2, sc2, p0);
  run_block(l2b, p0, 64, 1, y2, nullptr, p1);
  // layer3 -> @32, 256ch
  _Float16* y3  = (_Float16*)alloc(szf(256, 32));
  _Float16* sc3 = (_Float16*)alloc(szf(256, 32));
  _Float16* q0  = (_Float16*)alloc(szf(256, 32));
  _Float16* q1  = (_Float16*)alloc(szf(256, 32));
  run_block(l3a, p1, 64, 2, y3, sc3, q0);
  run_block(l3b, q0, 32, 1, y3, nullptr, q1);
  // layer4 -> @16, 512ch
  _Float16* y4  = (_Float16*)alloc(szf(512, 16));
  _Float16* sc4 = (_Float16*)alloc(szf(512, 16));
  _Float16* r0  = (_Float16*)alloc(szf(512, 16));
  _Float16* r1  = (_Float16*)alloc(szf(512, 16));
  run_block(l4a, q1, 32, 2, y4, sc4, r0);
  run_block(l4b, r0, 16, 1, y4, nullptr, r1);

  // SPP concat [x, mp5, mp9, mp13] -> 2048ch @16 (NHWC channel concat)
  _Float16* cc = (_Float16*)alloc((size_t)Bn * 2048 * 256 * sizeof(_Float16));
  { long n = (long)Bn * 512 * 256; int nb = (int)((n + 255) / 256);
    k_maxpool<<<nb, 256, 0, stream>>>(r1, cc, Bn, 512, 16, 16, 1, 1, 0, 16, 16, 2048, 0);
    k_maxpool<<<nb, 256, 0, stream>>>(r1, cc, Bn, 512, 16, 16, 5, 1, 2, 16, 16, 2048, 512);
    k_maxpool<<<nb, 256, 0, stream>>>(r1, cc, Bn, 512, 16, 16, 9, 1, 4, 16, 16, 2048, 1024);
    k_maxpool<<<nb, 256, 0, stream>>>(r1, cc, Bn, 512, 16, 16, 13, 1, 6, 16, 16, 2048, 1536); }
  _Float16* s1 = (_Float16*)alloc(szf(512, 16));
  conv(cc, spp, s1, nullptr, 16, 16, 1, 0, 2);                 // leaky
  // SAM
  _Float16* tt = (_Float16*)alloc(szf(512, 16));
  conv(s1, sam, tt, nullptr, 16, 16, 1, 0, 0);
  _Float16* s2 = (_Float16*)alloc(szf(512, 16));
  { long n = (long)Bn * 512 * 256;
    k_sam<<<(int)((n + 255) / 256), 256, 0, stream>>>(s1, tt, s2, n); }
  // head convs (leaky)
  _Float16* u0 = (_Float16*)alloc(szf(256, 16));
  conv(s2, cs[0], u0, nullptr, 16, 16, 1, 0, 2);
  _Float16* u1 = (_Float16*)alloc(szf(512, 16));
  conv(u0, cs[1], u1, nullptr, 16, 16, 1, 1, 2);
  _Float16* u2 = (_Float16*)alloc(szf(256, 16));
  conv(u1, cs[2], u2, nullptr, 16, 16, 1, 0, 2);
  _Float16* u3 = (_Float16*)alloc(szf(512, 16));
  conv(u2, cs[3], u3, nullptr, 16, 16, 1, 1, 2);
  // prediction (85 channels, scalar epilogue path)
  _Float16* pr = (_Float16*)alloc((size_t)Bn * 85 * 256 * sizeof(_Float16));
  conv(u3, pred, pr, nullptr, 16, 16, 1, 0, 0);

  // decode + NMS
  float* wscores = (float*)alloc((size_t)Bn * 256 * sizeof(float));
  float* outf = (float*)d_out;
  k_decode<<<(Bn * 256 + 255) / 256, 256, 0, stream>>>(pr, outf, wscores, Bn, 256, 16, 32.f);
  k_nms<<<Bn, 256, 0, stream>>>(outf, outf + (long)Bn * 256 * 5, wscores,
                                outf + (long)Bn * 256 * 4, 256);
}
